// MoE_1013612282293
// MI455X (gfx1250) — matrix operand order repaired and checked by reading, not compiled, not run on hardware
//
#include <hip/hip_runtime.h>
#include <math.h>

typedef __attribute__((ext_vector_type(16))) _Float16 v16h;
typedef __attribute__((ext_vector_type(8)))  _Float16 v8h;
typedef __attribute__((ext_vector_type(4)))  _Float16 v4h;
typedef __attribute__((ext_vector_type(8)))  float    v8f;
typedef __attribute__((ext_vector_type(4)))  float    v4f;

#define D_DIM 768
#define F_DIM 3072
#define E_NUM 8
#define FC    128                 // F chunk width processed per iteration
#define NCHUNK (F_DIM / FC)       // 24
#define MT    32                  // rows (tokens) per block: two 16-row subtiles

static __device__ __forceinline__ v8f zero8() {
  v8f z = {0.f, 0.f, 0.f, 0.f, 0.f, 0.f, 0.f, 0.f};
  return z;
}

// ---------------------------------------------------------------------------
// Kernel 0a: transpose + fp32->fp16 convert expert weights.
// w1[e]: [D,F] -> w1t[e]: [F,D] row-major. w2[e]: [F,D] -> w2t[e]: [D,F].
// ---------------------------------------------------------------------------
__global__ void transpose_cvt_kernel(const float* __restrict__ w1,
                                     const float* __restrict__ w2,
                                     _Float16* __restrict__ w1t,
                                     _Float16* __restrict__ w2t) {
  __shared__ _Float16 tile[64][65];
  const int e     = blockIdx.y;
  const int which = blockIdx.z;
  const size_t mat = (size_t)D_DIM * F_DIM;
  const float*  src;
  _Float16*     dst;
  int R, C;
  if (which == 0) { src = w1 + e * mat; dst = w1t + e * mat; R = D_DIM; C = F_DIM; }
  else            { src = w2 + e * mat; dst = w2t + e * mat; R = F_DIM; C = D_DIM; }
  const int tilesC = C >> 6;
  const int tr = blockIdx.x / tilesC;
  const int tc = blockIdx.x % tilesC;
  const int r0 = tr << 6, c0 = tc << 6;

  const int c  = threadIdx.x & 63;
  const int rb = threadIdx.x >> 6;
#pragma unroll
  for (int k = 0; k < 16; ++k) {
    int r = rb + (k << 2);
    tile[r][c] = (_Float16)src[(size_t)(r0 + r) * C + (c0 + c)];
  }
  __syncthreads();
  const int rr = threadIdx.x & 63;
  const int cb = threadIdx.x >> 6;
#pragma unroll
  for (int k = 0; k < 16; ++k) {
    int cc = cb + (k << 2);
    dst[(size_t)(c0 + cc) * R + (r0 + rr)] = tile[rr][cc];
  }
}

// ---------------------------------------------------------------------------
// Kernel 0b: convert activations fp32 -> fp16 (enables async fp16 row gather).
// ---------------------------------------------------------------------------
__global__ void x_cvt_kernel(const float* __restrict__ x,
                             _Float16* __restrict__ xh, int n) {
  int idx = (blockIdx.x * blockDim.x + threadIdx.x) * 4;
  if (idx >= n) return;
  v4f f = *(const v4f*)(x + idx);
  v4h h;
  h.x = (_Float16)f.x; h.y = (_Float16)f.y;
  h.z = (_Float16)f.z; h.w = (_Float16)f.w;
  *(v4h*)(xh + idx) = h;
}

// ---------------------------------------------------------------------------
// Kernel 1: router. One wave32 per token.
// ---------------------------------------------------------------------------
__global__ void router_kernel(const float* __restrict__ x,
                              const float* __restrict__ rw,
                              const float* __restrict__ rb,
                              int*   __restrict__ cnt,
                              int*   __restrict__ bidx,
                              float* __restrict__ bwgt,
                              int n_tok) {
  const int wave = (int)((blockIdx.x * blockDim.x + threadIdx.x) >> 5);
  const int lane = threadIdx.x & 31;
  if (wave >= n_tok) return;

  const float* xr = x + (size_t)wave * D_DIM;
  float acc[E_NUM];
#pragma unroll
  for (int e = 0; e < E_NUM; ++e) acc[e] = 0.f;

  for (int d = lane; d < D_DIM; d += 32) {
    const float xv = xr[d];
    const float* rwr = rw + (size_t)d * E_NUM;
#pragma unroll
    for (int e = 0; e < E_NUM; ++e) acc[e] += xv * rwr[e];
  }
#pragma unroll
  for (int off = 16; off >= 1; off >>= 1) {
#pragma unroll
    for (int e = 0; e < E_NUM; ++e) acc[e] += __shfl_xor(acc[e], off, 32);
  }
#pragma unroll
  for (int e = 0; e < E_NUM; ++e) acc[e] += rb[e];

  float mx = acc[0];
#pragma unroll
  for (int e = 1; e < E_NUM; ++e) mx = fmaxf(mx, acc[e]);
  float p[E_NUM];
#pragma unroll
  for (int e = 0; e < E_NUM; ++e) p[e] = __expf(acc[e] - mx);

  int i1 = 0; float p1 = p[0];
#pragma unroll
  for (int e = 1; e < E_NUM; ++e) { if (p[e] > p1) { p1 = p[e]; i1 = e; } }
  int i2 = (i1 == 0) ? 1 : 0; float p2 = p[i2];
#pragma unroll
  for (int e = 0; e < E_NUM; ++e) {
    if (e != i1 && p[e] > p2) { p2 = p[e]; i2 = e; }
  }
  const float inv = 1.0f / (p1 + p2);

  if (lane == 0) {
    int pos = atomicAdd(&cnt[i1], 1);
    bidx[i1 * n_tok + pos] = wave;
    bwgt[i1 * n_tok + pos] = p1 * inv;
    pos = atomicAdd(&cnt[i2], 1);
    bidx[i2 * n_tok + pos] = wave;
    bwgt[i2 * n_tok + pos] = p2 * inv;
  }
}

// ---------------------------------------------------------------------------
// Kernel 2: fused expert FFN for one (expert, 32-token tile).
// 256 threads = 8 waves. Wave w owns H columns [w*16,w*16+16) of each F-chunk
// and Y columns [w*96, w*96+96), each for two 16-row M-subtiles.
// ---------------------------------------------------------------------------
__global__ void __launch_bounds__(256, 1)
moe_ffn_kernel(const _Float16*  __restrict__ xh,
               const _Float16*  __restrict__ w1t,
               const _Float16*  __restrict__ w2t,
               const float*     __restrict__ b1,
               const float*     __restrict__ b2,
               const int*       __restrict__ cnt,
               const int*       __restrict__ bidx,
               const float*     __restrict__ bwgt,
               float*           __restrict__ out,
               int n_tok, int tiles_per_e) {
  __shared__ __align__(16) _Float16 sx[MT * D_DIM];   // 48 KB gathered x rows (fp16)
  __shared__ __align__(16) _Float16 sh[MT * FC];      // 8 KB  H chunk (fp16)
  __shared__ int   s_tok[MT];
  __shared__ float s_wgt[MT];

  const int e    = blockIdx.x / tiles_per_e;
  const int tile = blockIdx.x % tiles_per_e;
  const int ce   = cnt[e];
  const int row0 = tile * MT;
  if (row0 >= ce) return;                       // uniform: whole block exits
  const int valid = min(MT, ce - row0);

  const int tid  = threadIdx.x;
  const int wv   = tid >> 5;
  const int lane = tid & 31;
  const int lm   = lane & 15;   // A-frag row m / B-frag & C col n
  const int lh   = lane >> 4;   // K-group selector

  if (tid < MT) {
    if (tid < valid) {
      s_tok[tid] = bidx[e * n_tok + row0 + tid];
      s_wgt[tid] = bwgt[e * n_tok + row0 + tid];
    } else { s_tok[tid] = 0; s_wgt[tid] = 0.f; }
  }
  __syncthreads();

  // --- async gather of 32 fp16 token rows into LDS (ASYNCcnt path) ---
  {
    const uint32_t sx_base = (uint32_t)(uintptr_t)(void*)&sx[0];
#pragma unroll
    for (int r = 0; r < MT / 8; ++r) {          // 4 rows per wave
      const int m = wv * (MT / 8) + r;
      const _Float16* g = xh + (size_t)s_tok[m] * D_DIM;
#pragma unroll
      for (int piece = 0; piece < 3; ++piece) { // 3 x 512B per 1536B row
        uint32_t ldsoff = sx_base + (uint32_t)(m * (D_DIM * 2) + piece * 512 + lane * 16);
        const void* ga = (const void*)(g + piece * 256 + lane * 8);
        asm volatile("global_load_async_to_lds_b128 %0, %1, off"
                     :: "v"(ldsoff), "v"(ga) : "memory");
      }
    }
    asm volatile("s_wait_asynccnt 0x0" ::: "memory");
  }
  __syncthreads();

  const size_t matsz = (size_t)D_DIM * F_DIM;
  const _Float16* W1e = w1t + (size_t)e * matsz;  // [F,D] row-major
  const _Float16* W2e = w2t + (size_t)e * matsz;  // [D,F] row-major
  const float* b1e = b1 + e * F_DIM;
  const float* b2e = b2 + e * D_DIM;

  v8f yacc[12];                                   // [t][sub]: 6 n-tiles x 2 M-subtiles
#pragma unroll
  for (int t = 0; t < 12; ++t) yacc[t] = zero8();

  for (int chunk = 0; chunk < NCHUNK; ++chunk) {
    // ---- GEMM1: two 16x16 H subtiles for this wave, K = 768 ----
    v8f hc0 = zero8(), hc1 = zero8();
    const int fcol = chunk * FC + wv * 16 + lm;   // this lane's H/B column
    const _Float16* bB = W1e + (size_t)fcol * D_DIM;
#pragma unroll 4
    for (int kc = 0; kc < D_DIM / 32; ++kc) {     // 24 iters, B reused for 2 WMMA
      v16h a0, a1, bf;
      const _Float16* ap0 = sx + lm * D_DIM + kc * 32 + lh * 8;
      ((v8h*)&a0)[0] = *(const v8h*)(ap0);
      ((v8h*)&a0)[1] = *(const v8h*)(ap0 + 16);
      const _Float16* ap1 = ap0 + 16 * D_DIM;
      ((v8h*)&a1)[0] = *(const v8h*)(ap1);
      ((v8h*)&a1)[1] = *(const v8h*)(ap1 + 16);
      const _Float16* bp = bB + kc * 32 + lh * 8;
      ((v8h*)&bf)[0] = *(const v8h*)(bp);
      ((v8h*)&bf)[1] = *(const v8h*)(bp + 16);
      hc0 = __builtin_amdgcn_wmma_f32_16x16x32_f16(false, a0, false, bf,
                                                   (short)0, hc0, false, false);
      hc1 = __builtin_amdgcn_wmma_f32_16x16x32_f16(false, a1, false, bf,
                                                   (short)0, hc1, false, false);
    }
    // bias + exact GELU, pack both subtiles to LDS
    {
      const float bias = b1e[fcol];
#pragma unroll
      for (int j = 0; j < 8; ++j) {
        const int mm = lh * 8 + j;
        float v0 = hc0[j] + bias;
        float v1 = hc1[j] + bias;
        float g0 = 0.5f * v0 * (1.0f + erff(v0 * 0.70710678118f));
        float g1 = 0.5f * v1 * (1.0f + erff(v1 * 0.70710678118f));
        sh[mm * FC + wv * 16 + lm]        = (_Float16)g0;
        sh[(16 + mm) * FC + wv * 16 + lm] = (_Float16)g1;
      }
    }
    __syncthreads();

    // ---- GEMM2 partial: Y += H_chunk (32xFC) * W2^T chunk ----
#pragma unroll
    for (int kc = 0; kc < FC / 32; ++kc) {        // 4 iters
      v16h a0, a1;
      const _Float16* ap0 = sh + lm * FC + kc * 32 + lh * 8;
      ((v8h*)&a0)[0] = *(const v8h*)(ap0);
      ((v8h*)&a0)[1] = *(const v8h*)(ap0 + 16);
      const _Float16* ap1 = ap0 + 16 * FC;
      ((v8h*)&a1)[0] = *(const v8h*)(ap1);
      ((v8h*)&a1)[1] = *(const v8h*)(ap1 + 16);
#pragma unroll
      for (int t = 0; t < 6; ++t) {
        const int n = wv * 96 + t * 16 + lm;
        const _Float16* bp = W2e + (size_t)n * F_DIM + chunk * FC + kc * 32 + lh * 8;
        v16h bf;
        ((v8h*)&bf)[0] = *(const v8h*)(bp);
        ((v8h*)&bf)[1] = *(const v8h*)(bp + 16);
        yacc[2 * t]     = __builtin_amdgcn_wmma_f32_16x16x32_f16(false, a0, false, bf,
                                                                 (short)0, yacc[2 * t], false, false);
        yacc[2 * t + 1] = __builtin_amdgcn_wmma_f32_16x16x32_f16(false, a1, false, bf,
                                                                 (short)0, yacc[2 * t + 1], false, false);
      }
    }
    __syncthreads();
  }

  // ---- epilogue: bias2, gate weight, scatter-add ----
#pragma unroll
  for (int t = 0; t < 6; ++t) {
    const int n = wv * 96 + t * 16 + lm;
    const float bb = b2e[n];
#pragma unroll
    for (int sub = 0; sub < 2; ++sub) {
#pragma unroll
      for (int j = 0; j < 8; ++j) {
        const int mm = sub * 16 + lh * 8 + j;
        if (mm < valid) {
          const float val = (yacc[2 * t + sub][j] + bb) * s_wgt[mm];
          atomicAdd(&out[(size_t)s_tok[mm] * D_DIM + n], val);
        }
      }
    }
  }
}

// ---------------------------------------------------------------------------
extern "C" void kernel_launch(void* const* d_in, const int* in_sizes, int n_in,
                              void* d_out, int out_size, void* d_ws, size_t ws_size,
                              hipStream_t stream) {
  const float* x        = (const float*)d_in[0];
  const float* router_w = (const float*)d_in[1];
  const float* router_b = (const float*)d_in[2];
  const float* w1       = (const float*)d_in[3];
  const float* b1       = (const float*)d_in[4];
  const float* w2       = (const float*)d_in[5];
  const float* b2       = (const float*)d_in[6];
  float* out = (float*)d_out;

  const int n_tok = in_sizes[0] / D_DIM;          // B*S = 8192
  const int n_x   = in_sizes[0];

  // workspace layout
  char* ws = (char*)d_ws;
  const size_t szWT = (size_t)E_NUM * D_DIM * F_DIM * sizeof(_Float16); // 37.75 MB
  _Float16* w1t = (_Float16*)(ws);
  _Float16* w2t = (_Float16*)(ws + szWT);
  char* p = ws + 2 * szWT;
  int*   cnt  = (int*)p;                 p += 256;
  int*   bidx = (int*)p;                 p += (size_t)E_NUM * n_tok * sizeof(int);
  float* bwgt = (float*)p;               p += (size_t)E_NUM * n_tok * sizeof(float);
  _Float16* xh = (_Float16*)p;           // n_x halfs (12.6 MB)

  hipMemsetAsync(cnt, 0, 256, stream);
  hipMemsetAsync(out, 0, (size_t)n_tok * D_DIM * sizeof(float), stream);

  // 0a) weight transpose + fp16 convert
  dim3 tgrid((D_DIM / 64) * (F_DIM / 64), E_NUM, 2);
  transpose_cvt_kernel<<<tgrid, 256, 0, stream>>>(w1, w2, w1t, w2t);

  // 0b) activation fp32 -> fp16
  x_cvt_kernel<<<(n_x / 4 + 255) / 256, 256, 0, stream>>>(x, xh, n_x);

  // 1) router (one wave per token, 8 waves per block)
  router_kernel<<<(n_tok + 7) / 8, 256, 0, stream>>>(x, router_w, router_b,
                                                     cnt, bidx, bwgt, n_tok);

  // 2) fused expert FFN (worst-case grid; empty tiles exit immediately)
  const int tiles_per_e = (n_tok + MT - 1) / MT;
  moe_ffn_kernel<<<E_NUM * tiles_per_e, 256, 0, stream>>>(
      xh, w1t, w2t, b1, b2, cnt, bidx, bwgt, out, n_tok, tiles_per_e);
}
